// MultiHeadAttention_47459388620914
// MI455X (gfx1250) — compile-verified, matrix operand order checked by reading, not run on hardware
//
#include <hip/hip_runtime.h>
#include <hip/hip_bf16.h>

typedef __attribute__((ext_vector_type(16))) _Float16     v16h;
typedef __attribute__((ext_vector_type(8)))  _Float16     v8h;
typedef __attribute__((ext_vector_type(4)))  _Float16     v4h;
typedef __attribute__((ext_vector_type(8)))  float        v8f;
typedef __attribute__((ext_vector_type(4)))  float        v4f;
typedef __attribute__((ext_vector_type(4)))  unsigned int v4u;
typedef __attribute__((ext_vector_type(8)))  int          v8i;
typedef __attribute__((ext_vector_type(4)))  int          v4i;

#if defined(__has_builtin)
#if __has_builtin(__builtin_amdgcn_tensor_load_to_lds)
#define HAVE_TDM 1
#endif
#endif
#ifndef HAVE_TDM
#define HAVE_TDM 0
#endif

// ---------------------------------------------------------------------------
// Fragment helpers (CDNA5 WMMA 16x16x32 f16 layouts, wave32)
// A (16x32): lane m = lane&15, kbase=(lane>>4)*8; elems 0..7 = K kbase..+7,
//            elems 8..15 = K kbase+16..+23.  -> two 16B loads at p, p+16.
// ---------------------------------------------------------------------------
__device__ __forceinline__ v16h frag_ld(const _Float16* p) {
  v8h lo = *(const v8h*)(p);
  v8h hi = *(const v8h*)(p + 16);
  v16h r;
#pragma unroll
  for (int i = 0; i < 8; ++i) { r[i] = lo[i]; r[i + 8] = hi[i]; }
  return r;
}

__device__ __forceinline__ v8f wmma_f16(v16h a, v16h b, v8f c) {
  return __builtin_amdgcn_wmma_f32_16x16x32_f16(
      false, a, false, b, (short)0, c, false, false);
}

// ---------------------------------------------------------------------------
// f32 -> f16 convert (vectorized x4)
// ---------------------------------------------------------------------------
__global__ void cvt_f32_to_f16(const float* __restrict__ in,
                               _Float16* __restrict__ out, int n4) {
  int i = blockIdx.x * blockDim.x + threadIdx.x;
  if (i < n4) {
    v4f v = *(const v4f*)(in + (size_t)i * 4);
    v4h h;
#pragma unroll
    for (int j = 0; j < 4; ++j) h[j] = (_Float16)v[j];
    *(v4h*)(out + (size_t)i * 4) = h;
  }
}

// transpose [R x C] f32 -> [C x R] f16
__global__ void transpose_cvt(const float* __restrict__ in,
                              _Float16* __restrict__ out, int R, int C) {
  int i = blockIdx.x * blockDim.x + threadIdx.x;
  if (i < R * C) {
    int r = i / C, c = i % C;
    out[(size_t)c * R + r] = (_Float16)in[i];
  }
}

// ---------------------------------------------------------------------------
// WMMA GEMM: out[M,N] = A[M,K] * Bt[K,N] (+bias)
// WG = 128 threads (4 waves), tile 128(M) x 64(N), K-step 32, double-buffered
// LDS. Each wave computes 32x64 (2 A-frags x 4 B-frags = 8 WMMA / K-step).
// mode 0: f32 store.  mode 1: qkv split-scatter [B,H,T,d] f16, Q * 0.125.
// ---------------------------------------------------------------------------
#define TGM 128
#define TGN 64
#define TK  32

__global__ __launch_bounds__(128) void gemm_wmma_kernel(
    const _Float16* __restrict__ A, const _Float16* __restrict__ Bt,
    const float* __restrict__ bias, int M, int N, int K, int mode,
    float* __restrict__ outF, _Float16* __restrict__ q16,
    _Float16* __restrict__ k16, _Float16* __restrict__ v16) {
  __shared__ __align__(16) _Float16 ldsA[2][TGM * TK];  // [row][k]
  __shared__ __align__(16) _Float16 ldsB[2][TGN * TK];  // [n][k] (transposed)

  const int tid   = threadIdx.x;
  const int wave  = tid >> 5;
  const int lane  = tid & 31;
  const int lhalf = lane & 15;
  const int kbase = (lane >> 4) * 8;
  const int m0    = blockIdx.y * TGM;
  const int n0    = blockIdx.x * TGN;

  v8f acc[2][4] = {};

  const int bk = tid >> 2, bnseg = (tid & 3) * 16;

  auto load_tile = [&](int buf, int k0) {
    // A tile 128x32: each thread loads one full row (64B = 4 x b128)
    const _Float16* srcA = A + (size_t)(m0 + tid) * K + k0;
    v8h a0 = *(const v8h*)(srcA);
    v8h a1 = *(const v8h*)(srcA + 8);
    v8h a2 = *(const v8h*)(srcA + 16);
    v8h a3 = *(const v8h*)(srcA + 24);
    _Float16* dA = &ldsA[buf][tid * TK];
    *(v8h*)(dA)      = a0;
    *(v8h*)(dA + 8)  = a1;
    *(v8h*)(dA + 16) = a2;
    *(v8h*)(dA + 24) = a3;
    // B tile 32x64 -> LDS as [n][k]
    const _Float16* srcB = Bt + (size_t)(k0 + bk) * N + n0 + bnseg;
    v8h b0 = *(const v8h*)srcB;
    v8h b1 = *(const v8h*)(srcB + 8);
#pragma unroll
    for (int j = 0; j < 8; ++j) {
      ldsB[buf][(bnseg + j)     * TK + bk] = b0[j];
      ldsB[buf][(bnseg + 8 + j) * TK + bk] = b1[j];
    }
    __builtin_prefetch(srcA + TK, 0, 1);
    __builtin_prefetch(srcB + (size_t)TK * N, 0, 1);
  };

  load_tile(0, 0);
  __syncthreads();

  const int nk = K / TK;
  for (int kt = 0; kt < nk; ++kt) {
    const int cur = kt & 1;
    if (kt + 1 < nk) load_tile(cur ^ 1, (kt + 1) * TK);

    // preload ALL fragments, then back-to-back WMMAs
    v16h af[2], bf[4];
#pragma unroll
    for (int mi = 0; mi < 2; ++mi)
      af[mi] = frag_ld(&ldsA[cur][(wave * 32 + mi * 16 + lhalf) * TK + kbase]);
#pragma unroll
    for (int j = 0; j < 4; ++j)
      bf[j] = frag_ld(&ldsB[cur][(j * 16 + lhalf) * TK + kbase]);
#pragma unroll
    for (int mi = 0; mi < 2; ++mi)
#pragma unroll
      for (int j = 0; j < 4; ++j)
        acc[mi][j] = wmma_f16(af[mi], bf[j], acc[mi][j]);
    __syncthreads();
  }

  // epilogue
  const int rbase = (lane >> 4) * 8;
#pragma unroll
  for (int mi = 0; mi < 2; ++mi) {
#pragma unroll
    for (int j = 0; j < 4; ++j) {
      const int n  = n0 + j * 16 + lhalf;
      const float bv = bias[n];
#pragma unroll
      for (int i = 0; i < 8; ++i) {
        const int m   = m0 + wave * 32 + mi * 16 + rbase + i;
        const float v = acc[mi][j][i] + bv;
        if (mode == 0) {
          outF[(size_t)m * N + n] = v;
        } else {
          const int which = n >> 10;
          const int c = n & 1023;
          const int h = c >> 6, d = c & 63;
          const int b = m >> 11, t = m & 2047;
          const size_t idx = (((size_t)(b * 16 + h)) * 2048 + t) * 64 + d;
          if (which == 0)      q16[idx] = (_Float16)(v * 0.125f);
          else if (which == 1) k16[idx] = (_Float16)v;
          else                 v16[idx] = (_Float16)v;
        }
      }
    }
  }
}

// ---------------------------------------------------------------------------
// Flash attention: grid = (T/64 q-blocks, B*H).  128 threads = 4 waves,
// each wave owns 16 query rows. Key blocks of 32, online softmax.
// K tile is fetched by the Tensor Data Mover (wave 0 issues 2D-tile D#).
// ---------------------------------------------------------------------------
#define FA_KB 32

__global__ __launch_bounds__(128) void flash_attn_kernel(
    const _Float16* __restrict__ q16, const _Float16* __restrict__ k16,
    const _Float16* __restrict__ v16, _Float16* __restrict__ ctx16) {
  __shared__ __align__(16) _Float16 Kt[FA_KB * 64];   // [key][d]
  __shared__ __align__(16) _Float16 VtT[64 * FA_KB];  // [d][key]
  __shared__ __align__(16) float    Sw[4][16 * 32];
  __shared__ __align__(16) _Float16 Pw[4][16 * 32];
  __shared__ float rowM[4][16], rowL[4][16], rowC[4][16], rowInv[4][16];

  const int tid   = threadIdx.x;
  const int wave  = tid >> 5;
  const int lane  = tid & 31;
  const int lhalf = lane & 15;
  const int kbase = (lane >> 4) * 8;
  const int bh    = blockIdx.y;                 // b*16 + h
  const int qb    = blockIdx.x;
  const size_t headOff = (size_t)bh * 2048 * 64;
  const int t0 = qb * 64 + wave * 16;

  // preload Q A-fragments (16 rows x 64 d -> two K=32 chunks)
  v16h aq[2];
#pragma unroll
  for (int c = 0; c < 2; ++c)
    aq[c] = frag_ld(q16 + headOff + (size_t)(t0 + lhalf) * 64 + c * 32 + kbase);

  v8f acc[4] = {};
  if (lane < 16) { rowM[wave][lane] = -3.0e38f; rowL[wave][lane] = 0.f; }
  __syncthreads();

  const int key  = (tid * 16) >> 6;   // cooperative V-tile mapping
  const int dseg = (tid * 16) & 63;

  for (int kb = 0; kb < 2048 / FA_KB; ++kb) {
#if HAVE_TDM
    // ---- K tile [32 x 64] via Tensor Data Mover (2D tile descriptor) ----
    if (wave == 0) {
      const unsigned long long ga =
          (unsigned long long)(size_t)(k16 + headOff + (size_t)kb * FA_KB * 64);
      v4u g0;
      g0[0] = 1u;                                     // count=1, user mode
      g0[1] = (unsigned)(size_t)(&Kt[0]);             // lds_addr (low 32 bits)
      g0[2] = (unsigned)(ga & 0xFFFFFFFFu);           // global_addr[31:0]
      g0[3] = (unsigned)((ga >> 32) & 0x01FFFFFFu)    // global_addr[56:32]
              | (2u << 30);                           // type = 2 (image)
      v8i g1;
      g1[0] = (int)(1u << 16);        // data_size = 1 (2 bytes/elem)
      g1[1] = (int)(64u << 16);       // tensor_dim0 = 64 (low 16 in [63:48])
      g1[2] = (int)(2048u << 16);     // tensor_dim1 = 2048 (low 16 in [95:80])
      g1[3] = (int)(64u << 16);       // tile_dim0 = 64 in [127:112]
      g1[4] = 32;                     // tile_dim1 = 32, tile_dim2 = 0
      g1[5] = 64;                     // tensor_dim0_stride = 64
      g1[6] = 0;                      // stride0 hi | stride1 lo
      g1[7] = 0;                      // stride1 hi
      v4i z4 = {0, 0, 0, 0};
      v8i z8 = {0, 0, 0, 0, 0, 0, 0, 0};
      __builtin_amdgcn_tensor_load_to_lds(g0, g1, z4, z4, z8, 0);
      __builtin_amdgcn_s_wait_tensorcnt(0);
    }
#else
    {
      const _Float16* srcK = k16 + headOff + (size_t)(kb * FA_KB + key) * 64 + dseg;
      v8h ka = *(const v8h*)srcK;
      v8h kb2 = *(const v8h*)(srcK + 8);
      *(v8h*)&Kt[key * 64 + dseg]     = ka;
      *(v8h*)&Kt[key * 64 + dseg + 8] = kb2;
    }
#endif
    // ---- V tile transposed [64 x 32] (cooperative) ----
    {
      const _Float16* srcV = v16 + headOff + (size_t)(kb * FA_KB + key) * 64 + dseg;
      v8h va = *(const v8h*)srcV;
      v8h vb = *(const v8h*)(srcV + 8);
#pragma unroll
      for (int j = 0; j < 8; ++j) {
        VtT[(dseg + j)     * FA_KB + key] = va[j];
        VtT[(dseg + 8 + j) * FA_KB + key] = vb[j];
      }
    }
    __syncthreads();

    // --- S = Q K^T : preload all 4 B-fragments, then 4 WMMAs ---
    v16h bkf[2][2];
#pragma unroll
    for (int j = 0; j < 2; ++j)
#pragma unroll
      for (int c = 0; c < 2; ++c)
        bkf[j][c] = frag_ld(&Kt[(j * 16 + lhalf) * 64 + c * 32 + kbase]);
    v8f sacc[2] = {};
#pragma unroll
    for (int j = 0; j < 2; ++j)
#pragma unroll
      for (int c = 0; c < 2; ++c)
        sacc[j] = wmma_f16(aq[c], bkf[j][c], sacc[j]);

    // spill S to per-wave LDS for softmax
#pragma unroll
    for (int j = 0; j < 2; ++j)
#pragma unroll
      for (int i = 0; i < 8; ++i)
        Sw[wave][(i + (lane >> 4) * 8) * 32 + j * 16 + lhalf] = sacc[j][i];
    __syncthreads();

    // --- online softmax: lanes 0..15 each own one row ---
    if (lane < 16) {
      const float* srow = &Sw[wave][lane * 32];
      float mold = rowM[wave][lane];
      float mnew = mold;
#pragma unroll
      for (int j = 0; j < 32; ++j) mnew = fmaxf(mnew, srow[j]);
      float corr = __expf(mold - mnew);
      float lsum = 0.f;
#pragma unroll
      for (int j = 0; j < 32; ++j) {
        float p = __expf(srow[j] - mnew);
        lsum += p;
        Pw[wave][lane * 32 + j] = (_Float16)p;
      }
      rowM[wave][lane] = mnew;
      rowL[wave][lane] = rowL[wave][lane] * corr + lsum;
      rowC[wave][lane] = corr;
    }
    __syncthreads();

    // rescale running context accumulators
#pragma unroll
    for (int i = 0; i < 8; ++i) {
      float cf = rowC[wave][i + (lane >> 4) * 8];
#pragma unroll
      for (int j = 0; j < 4; ++j) acc[j][i] *= cf;
    }

    // --- ctx += P * V : preload fragments, then 4 WMMAs ---
    v16h ap = frag_ld(&Pw[wave][lhalf * 32 + kbase]);
    v16h bvf[4];
#pragma unroll
    for (int j = 0; j < 4; ++j)
      bvf[j] = frag_ld(&VtT[(j * 16 + lhalf) * FA_KB + kbase]);
#pragma unroll
    for (int j = 0; j < 4; ++j)
      acc[j] = wmma_f16(ap, bvf[j], acc[j]);
    __syncthreads();
  }

  if (lane < 16) rowInv[wave][lane] = 1.0f / rowL[wave][lane];
  __syncthreads();

  const int b = bh >> 4, h = bh & 15;
#pragma unroll
  for (int i = 0; i < 8; ++i) {
    const int row = i + (lane >> 4) * 8;
    const float inv = rowInv[wave][row];
    const int t = t0 + row;
#pragma unroll
    for (int j = 0; j < 4; ++j) {
      const int d = j * 16 + lhalf;
      ctx16[((size_t)(b * 2048 + t)) * 1024 + h * 64 + d] =
          (_Float16)(acc[j][i] * inv);
    }
  }
}

// ---------------------------------------------------------------------------
// Host-side launch
// ---------------------------------------------------------------------------
extern "C" void kernel_launch(void* const* d_in, const int* in_sizes, int n_in,
                              void* d_out, int out_size, void* d_ws,
                              size_t ws_size, hipStream_t stream) {
  const float* x     = (const float*)d_in[0];   // [2,2048,1024]
  const float* W_qkv = (const float*)d_in[1];   // [3072,1024]
  const float* b_qkv = (const float*)d_in[2];   // [3072]
  const float* W_out = (const float*)d_in[3];   // [1024,1024]
  const float* b_out = (const float*)d_in[4];   // [1024]
  float* out = (float*)d_out;                   // [2,2048,1024]

  char* ws = (char*)d_ws;
  _Float16* x16   = (_Float16*)(ws);                        //  8 MB
  _Float16* WqkvT = (_Float16*)(ws + (size_t)(8  << 20));   //  6 MB  [1024 x 3072]
  _Float16* WoutT = (_Float16*)(ws + (size_t)(14 << 20));   //  2 MB  [1024 x 1024]
  _Float16* q16   = (_Float16*)(ws + (size_t)(16 << 20));   //  8 MB  [B,H,T,d]
  _Float16* k16   = (_Float16*)(ws + (size_t)(24 << 20));   //  8 MB
  _Float16* v16   = (_Float16*)(ws + (size_t)(32 << 20));   //  8 MB
  _Float16* ctx16 = (_Float16*)(ws + (size_t)(40 << 20));   //  8 MB  [B,T,C]

  const int NTOK = 4096;   // B*T
  const int C    = 1024;

  cvt_f32_to_f16<<<(NTOK * C / 4 + 255) / 256, 256, 0, stream>>>(x, x16, NTOK * C / 4);
  transpose_cvt<<<(3 * C * C + 255) / 256, 256, 0, stream>>>(W_qkv, WqkvT, 3 * C, C);
  transpose_cvt<<<(C * C + 255) / 256, 256, 0, stream>>>(W_out, WoutT, C, C);

  gemm_wmma_kernel<<<dim3(3 * C / TGN, NTOK / TGM), 128, 0, stream>>>(
      x16, WqkvT, b_qkv, NTOK, 3 * C, C, /*mode=*/1, nullptr, q16, k16, v16);

  flash_attn_kernel<<<dim3(2048 / 64, 32), 128, 0, stream>>>(q16, k16, v16, ctx16);

  gemm_wmma_kernel<<<dim3(C / TGN, NTOK / TGM), 128, 0, stream>>>(
      ctx16, WoutT, b_out, NTOK, C, C, /*mode=*/0, out, nullptr, nullptr, nullptr);
}
